// SABlock_54537494725114
// MI455X (gfx1250) — matrix operand order repaired and checked by reading, not compiled, not run on hardware
//
#include <hip/hip_runtime.h>
#include <hip/hip_bf16.h>

typedef float  v2f   __attribute__((ext_vector_type(2)));
typedef float  v8f   __attribute__((ext_vector_type(8)));
typedef __bf16 v4bf  __attribute__((ext_vector_type(4)));
typedef __bf16 v8bf  __attribute__((ext_vector_type(8)));
typedef __bf16 v16bf __attribute__((ext_vector_type(16)));

__device__ __forceinline__ v8f wmma_f32(v2f a, v2f b, v8f c) {
    return __builtin_amdgcn_wmma_f32_16x16x4_f32(false, a, false, b, (short)0, c,
                                                 false, false);
}
__device__ __forceinline__ v8f wmma_bf16(v16bf a, v16bf b, v8f c) {
    return __builtin_amdgcn_wmma_f32_16x16x32_bf16(false, a, false, b, (short)0, c,
                                                   false, false);
}

// A-fragment (16-bit 16x32): elems 0..7 = K[k0+8*hw .. +7], elems 8..15 = K[k0+16+8*hw .. +7]
__device__ __forceinline__ v16bf load_a_frag(const __bf16* __restrict__ row, int k0, int hw) {
    v8bf a = *(const v8bf*)(row + k0 + 8 * hw);
    v8bf b = *(const v8bf*)(row + k0 + 16 + 8 * hw);
    return __builtin_shufflevector(a, b, 0, 1, 2, 3, 4, 5, 6, 7,
                                         8, 9, 10, 11, 12, 13, 14, 15);
}

// ---------------------------------------------------------------------------
// Elementwise split: hi = bf16(x), lo = bf16(x - hi).  n divisible by 1024.
// ---------------------------------------------------------------------------
__global__ __launch_bounds__(256) void decompose_bf16(const float* __restrict__ in,
                                                      __bf16* __restrict__ hi,
                                                      __bf16* __restrict__ lo, int n) {
    const int i = (blockIdx.x * 256 + threadIdx.x) * 4;
    if (i >= n) return;
    float4 f = *(const float4*)(in + i);
    __bf16 h0 = (__bf16)f.x, h1 = (__bf16)f.y, h2 = (__bf16)f.z, h3 = (__bf16)f.w;
    v4bf hv = {h0, h1, h2, h3};
    v4bf lv = {(__bf16)(f.x - (float)h0), (__bf16)(f.y - (float)h1),
               (__bf16)(f.z - (float)h2), (__bf16)(f.w - (float)h3)};
    *(v4bf*)(hi + i) = hv;
    *(v4bf*)(lo + i) = lv;
}

// ---------------------------------------------------------------------------
// C[M,N] = (Ah+Al)[M,K] @ (Bh+Bl)[N,K]^T  via  AhBh + AhBl + AlBh  (+ bias)
// block = 256 threads = 8 waves (2 in M x 4 in N); wave tile 32x64; block 64x256
// ---------------------------------------------------------------------------
template <bool ADD_BIAS>
__global__ __launch_bounds__(256) void gemm_nt_bf16x3(
    const __bf16* __restrict__ Ah, const __bf16* __restrict__ Al,
    const __bf16* __restrict__ Bh, const __bf16* __restrict__ Bl,
    const float* __restrict__ bias, float* __restrict__ C, int M, int N, int K) {
    const int lane = threadIdx.x & 31;
    const int wave = threadIdx.x >> 5;
    const int m0 = blockIdx.y * 64 + (wave & 1) * 32;
    const int n0 = blockIdx.x * 256 + (wave >> 1) * 64;

    const int fm  = lane & 15;
    const int hw  = lane >> 4;
    const int chi = hw * 8;

    v8f acc[2][4] = {};

    const __bf16* Ah0 = Ah + (size_t)(m0 + fm) * K;
    const __bf16* Al0 = Al + (size_t)(m0 + fm) * K;
    const __bf16* Ah1 = Ah + (size_t)(m0 + 16 + fm) * K;
    const __bf16* Al1 = Al + (size_t)(m0 + 16 + fm) * K;
    const __bf16* Bhr[4];
    const __bf16* Blr[4];
#pragma unroll
    for (int j = 0; j < 4; ++j) {
        Bhr[j] = Bh + (size_t)(n0 + 16 * j + fm) * K;
        Blr[j] = Bl + (size_t)(n0 + 16 * j + fm) * K;
    }

    for (int k0 = 0; k0 < K; k0 += 32) {
        v16bf ah0 = load_a_frag(Ah0, k0, hw);
        v16bf al0 = load_a_frag(Al0, k0, hw);
        v16bf ah1 = load_a_frag(Ah1, k0, hw);
        v16bf al1 = load_a_frag(Al1, k0, hw);
#pragma unroll
        for (int j = 0; j < 4; ++j) {
            v16bf bh = load_a_frag(Bhr[j], k0, hw);
            v16bf bl = load_a_frag(Blr[j], k0, hw);
            acc[0][j] = wmma_bf16(ah0, bh, acc[0][j]);
            acc[0][j] = wmma_bf16(ah0, bl, acc[0][j]);
            acc[0][j] = wmma_bf16(al0, bh, acc[0][j]);
            acc[1][j] = wmma_bf16(ah1, bh, acc[1][j]);
            acc[1][j] = wmma_bf16(ah1, bl, acc[1][j]);
            acc[1][j] = wmma_bf16(al1, bh, acc[1][j]);
        }
    }

#pragma unroll
    for (int i = 0; i < 2; ++i)
#pragma unroll
        for (int j = 0; j < 4; ++j) {
            const int col = n0 + 16 * j + fm;
            const float bv = ADD_BIAS ? bias[col] : 0.0f;
#pragma unroll
            for (int r = 0; r < 8; ++r)
                C[(size_t)(m0 + 16 * i + chi + r) * N + col] = acc[i][j][r] + bv;
        }
}

// ---------------------------------------------------------------------------
// Attention: exact fp32 WMMA path; writes bf16 hi/lo output for the next GEMM.
// ---------------------------------------------------------------------------
#define LDS_STRIDE 1028

__global__ __launch_bounds__(256) void attention(const float* __restrict__ qkv,
                                                 __bf16* __restrict__ oh,
                                                 __bf16* __restrict__ ol) {
    extern __shared__ float sc[];  // 32 * 1028 floats = 131584 bytes

    const int bh = blockIdx.y;
    const int b  = bh >> 4;
    const int h  = bh & 15;
    const int r0 = blockIdx.x * 32;

    const int lane = threadIdx.x & 31;
    const int wave = threadIdx.x >> 5;
    const int fm   = lane & 15;
    const int kb   = (lane >> 4) * 2;
    const int chi  = (lane >> 4) * 8;

    const size_t rs = 3 * 1024;
    const float* Qb = qkv + ((size_t)b * 1024 + r0) * rs + h * 64;
    const float* Kb = qkv + (size_t)b * 1024 * rs + 1024 + h * 64;
    const float* Vb = qkv + (size_t)b * 1024 * rs + 2048 + h * 64;

    v2f qf[2][16];
#pragma unroll
    for (int ms = 0; ms < 2; ++ms) {
        const float* qrow = Qb + (size_t)(ms * 16 + fm) * rs;
#pragma unroll
        for (int ks = 0; ks < 16; ++ks)
            qf[ms][ks] = *(const v2f*)(qrow + 4 * ks + kb);
    }

    const float scale = 0.125f;  // 1/sqrt(64)

    for (int t = 0; t < 8; ++t) {
        const int key0 = wave * 128 + t * 16;
        v8f acc0 = {}, acc1 = {};
        const float* krow = Kb + (size_t)(key0 + fm) * rs;
#pragma unroll
        for (int ks = 0; ks < 16; ++ks) {
            v2f bf = *(const v2f*)(krow + 4 * ks + kb);
            acc0 = wmma_f32(qf[0][ks], bf, acc0);
            acc1 = wmma_f32(qf[1][ks], bf, acc1);
        }
#pragma unroll
        for (int r = 0; r < 8; ++r) {
            sc[(chi + r) * LDS_STRIDE + key0 + fm]      = acc0[r] * scale;
            sc[(16 + chi + r) * LDS_STRIDE + key0 + fm] = acc1[r] * scale;
        }
    }
    __syncthreads();

    {
        const int row = threadIdx.x >> 3;
        const int seg = threadIdx.x & 7;
        float* srow = sc + row * LDS_STRIDE;
        float mx = -1e30f;
        for (int c = seg * 128; c < seg * 128 + 128; ++c) mx = fmaxf(mx, srow[c]);
#pragma unroll
        for (int off = 1; off < 8; off <<= 1) mx = fmaxf(mx, __shfl_xor(mx, off, 32));
        float sum = 0.0f;
        for (int c = seg * 128; c < seg * 128 + 128; ++c) {
            float e = __expf(srow[c] - mx);
            srow[c] = e;
            sum += e;
        }
#pragma unroll
        for (int off = 1; off < 8; off <<= 1) sum += __shfl_xor(sum, off, 32);
        const float inv = 1.0f / sum;
        for (int c = seg * 128; c < seg * 128 + 128; ++c) srow[c] *= inv;
    }
    __syncthreads();

    {
        const int ms = wave & 1;
        const int nt = wave >> 1;
        v8f acc = {};
        const float* vcol = Vb + nt * 16 + fm;
        const float* prow = sc + (size_t)(ms * 16 + fm) * LDS_STRIDE;
#pragma unroll 4
        for (int ks = 0; ks < 256; ++ks) {
            const int k0 = 4 * ks + kb;
            v2f af = *(const v2f*)(prow + k0);
            v2f bf;
            bf.x = vcol[(size_t)k0 * rs];
            bf.y = vcol[(size_t)(k0 + 1) * rs];
            acc = wmma_f32(af, bf, acc);
        }
        __bf16* obh = oh + ((size_t)b * 1024 + r0) * 1024 + h * 64;
        __bf16* obl = ol + ((size_t)b * 1024 + r0) * 1024 + h * 64;
#pragma unroll
        for (int r = 0; r < 8; ++r) {
            const int row = ms * 16 + chi + r;
            const float v = acc[r];
            const __bf16 hv = (__bf16)v;
            obh[(size_t)row * 1024 + nt * 16 + fm] = hv;
            obl[(size_t)row * 1024 + nt * 16 + fm] = (__bf16)(v - (float)hv);
        }
    }
}

// ---------------------------------------------------------------------------
extern "C" void kernel_launch(void* const* d_in, const int* in_sizes, int n_in,
                              void* d_out, int out_size, void* d_ws, size_t ws_size,
                              hipStream_t stream) {
    const float* x     = (const float*)d_in[0];  // (4,1024,1024)
    const float* w_qkv = (const float*)d_in[1];  // (3072,1024)
    const float* w_out = (const float*)d_in[2];  // (1024,1024)
    const float* b_out = (const float*)d_in[3];  // (1024)
    float* out = (float*)d_out;                  // (4,1024,1024)

    const size_t nX  = (size_t)4 * 1024 * 1024;
    const size_t nWq = (size_t)3072 * 1024;
    const size_t nWo = (size_t)1024 * 1024;

    char* p = (char*)d_ws;
    float*  qkv = (float*)p;  p += (size_t)4 * 1024 * 3072 * sizeof(float);
    __bf16* xh  = (__bf16*)p; p += nX * 2;
    __bf16* xl  = (__bf16*)p; p += nX * 2;
    __bf16* wqh = (__bf16*)p; p += nWq * 2;
    __bf16* wql = (__bf16*)p; p += nWq * 2;
    __bf16* woh = (__bf16*)p; p += nWo * 2;
    __bf16* wol = (__bf16*)p; p += nWo * 2;
    __bf16* ath = (__bf16*)p; p += nX * 2;
    __bf16* atl = (__bf16*)p; p += nX * 2;

    const int M = 4 * 1024;

    decompose_bf16<<<dim3(nX / (256 * 4)),  256, 0, stream>>>(x, xh, xl, (int)nX);
    decompose_bf16<<<dim3(nWq / (256 * 4)), 256, 0, stream>>>(w_qkv, wqh, wql, (int)nWq);
    decompose_bf16<<<dim3(nWo / (256 * 4)), 256, 0, stream>>>(w_out, woh, wol, (int)nWo);

    // 1) qkv = x @ w_qkv^T  (bf16x3, f32 accumulate): M=4096, N=3072, K=1024
    gemm_nt_bf16x3<false><<<dim3(3072 / 256, M / 64), 256, 0, stream>>>(
        xh, xl, wqh, wql, nullptr, qkv, M, 3072, 1024);

    // 2) attention (exact fp32 WMMA), emits bf16 hi/lo activations
    attention<<<dim3(1024 / 32, 4 * 16), 256, 32 * LDS_STRIDE * sizeof(float), stream>>>(
        qkv, ath, atl);

    // 3) out = attn @ w_out^T + b_out  (bf16x3): M=4096, N=1024, K=1024
    gemm_nt_bf16x3<true><<<dim3(1024 / 256, M / 64), 256, 0, stream>>>(
        ath, atl, woh, wol, b_out, out, M, 1024, 1024);
}
